// UA_OATD_70153995813557
// MI455X (gfx1250) — matrix operand order repaired and checked by reading, not compiled, not run on hardware
//
#include <hip/hip_runtime.h>
#include <hip/hip_bf16.h>

#define V_TOK 20000
#define E_DIM 128
#define H_DIM 256
#define H3_DIM 768
#define B_DIM 16
#define T_DIM 256
#define NE_DIM 640000

typedef __attribute__((ext_vector_type(16))) _Float16 v16h;
typedef __attribute__((ext_vector_type(8)))  float    v8f;
typedef __attribute__((ext_vector_type(4)))  unsigned int v4u;

union HFrag { v4u u[2]; v16h h; };

// ---------------- blocked WMMA GEMM: C[M,N] = A[M,K] * W[N,K]^T (+bias) ----
// A, W f16 row-major; C f32. Each wave computes a 16 x (16*NB) strip:
// the A fragment is loaded once per k-chunk and reused across NB WMMAs.
// Covers N-tile range [tnBase/16, tnBase/16 + tilesNJob); tilesNJob % NB == 0.
template <int NB>
__global__ void wmma_gemm_ab_t(const _Float16* __restrict__ A,
                               const _Float16* __restrict__ W,
                               const float* __restrict__ bias,
                               float* __restrict__ C,
                               int M, int N, int K,
                               int tnBase, int tilesNJob, int useBias)
{
    int wid  = (int)((blockIdx.x * blockDim.x + threadIdx.x) >> 5);
    int lane = threadIdx.x & 31;
    int jobsN = tilesNJob / NB;
    int totalJobs = (M >> 4) * jobsN;
    if (wid >= totalJobs) return;          // wave-uniform exit (EXEC all-1s for WMMA)
    int tm  = (wid / jobsN) << 4;
    int tn0 = tnBase + (wid % jobsN) * (NB * 16);
    int r15 = lane & 15;
    int hi  = lane >> 4;

    const _Float16* aRow = A + (size_t)(tm + r15) * (size_t)K;
    const _Float16* wRow0 = W + (size_t)(tn0 + r15) * (size_t)K;

    v8f acc[NB];
#pragma unroll
    for (int i = 0; i < NB; ++i) acc[i] = (v8f){};

    for (int k0 = 0; k0 < K; k0 += 32) {
        HFrag a;
        // A frag: lane<16 -> K{k0..k0+7, k0+16..k0+23}; lane>=16 -> +8
        a.u[0] = *(const v4u*)(aRow + k0 + hi * 8);
        a.u[1] = *(const v4u*)(aRow + k0 + 16 + hi * 8);
        if (k0 + 32 < K) __builtin_prefetch(aRow + k0 + 32, 0, 1);
#pragma unroll
        for (int nb = 0; nb < NB; ++nb) {
            const _Float16* wRow = wRow0 + (size_t)(nb * 16) * (size_t)K;
            HFrag b;
            // B frag: lane<16 -> K{k0..k0+7, k0+16..k0+23}; lane>=16 -> +8
            b.u[0] = *(const v4u*)(wRow + k0 + hi * 8);
            b.u[1] = *(const v4u*)(wRow + k0 + 16 + hi * 8);
            acc[nb] = __builtin_amdgcn_wmma_f32_16x16x32_f16(
                false, a.h, false, b.h, (short)0, acc[nb], false, false);
        }
    }
#pragma unroll
    for (int nb = 0; nb < NB; ++nb) {
        int col = tn0 + nb * 16 + r15;
        float bv = useBias ? bias[col] : 0.0f;
#pragma unroll
        for (int r = 0; r < 8; ++r) {
            int row = tm + hi * 8 + r;
            C[(size_t)row * (size_t)N + col] = acc[nb][r] + bv;
        }
    }
}

// ---------------- small helper kernels --------------------------------------
__global__ void fill_zero_f32(float* p, long n) {
    long i = (long)blockIdx.x * blockDim.x + threadIdx.x;
    for (; i < n; i += (long)gridDim.x * blockDim.x) p[i] = 0.0f;
}

__global__ void cvt_f32_to_f16(const float* __restrict__ src, _Float16* __restrict__ dst, long n) {
    long i = (long)blockIdx.x * blockDim.x + threadIdx.x;
    if (i < n) dst[i] = (_Float16)src[i];
}

// dst[c*R + r] = src[r*C + c]   (f32 -> f32)
__global__ void transpose_f32(const float* __restrict__ src, float* __restrict__ dst,
                              int R, int C) {
    long i = (long)blockIdx.x * blockDim.x + threadIdx.x;
    if (i >= (long)R * C) return;
    int r = (int)(i / C), c = (int)(i % C);
    dst[(size_t)c * R + r] = src[i];
}

__global__ void transpose_vp_f16(const float* __restrict__ Vp, _Float16* __restrict__ VpT) {
    int i = blockIdx.x * blockDim.x + threadIdx.x;       // 128*128
    int k = i >> 7, n = i & 127;
    VpT[(size_t)n * E_DIM + k] = (_Float16)Vp[(size_t)k * E_DIM + n];
}

// ---------------- SpMM: agg[row] += (d[row]*ev*d[col]) * emb[col] ----------
__global__ void spmm_scatter(const int* __restrict__ erow, const int* __restrict__ ecol,
                             const float* __restrict__ eval, const float* __restrict__ d,
                             const float* __restrict__ emb, float* __restrict__ agg)
{
    long idx = (long)blockIdx.x * blockDim.x + threadIdx.x;  // NE * 128
    long e = idx >> 7;
    int  j = (int)(idx & 127);
    if (e >= NE_DIM) return;
    int r = erow[e], c = ecol[e];
    float v = d[r] * eval[e] * d[c];
    atomicAdd(&agg[(size_t)r * E_DIM + j], v * emb[(size_t)c * E_DIM + j]);
}

// ---------------- gather e_input (and shifted decoder input) to f16 --------
__global__ void gather_inputs(const float* __restrict__ e_nodes, const int* __restrict__ trajs,
                              _Float16* __restrict__ eH, _Float16* __restrict__ dH)
{
    long idx = (long)blockIdx.x * blockDim.x + threadIdx.x;  // B*T*E
    long bt = idx >> 7;
    int  j  = (int)(idx & 127);
    if (bt >= (long)B_DIM * T_DIM) return;
    int t = (int)(bt % T_DIM);
    int tok = trajs[bt];
    eH[idx] = (_Float16)e_nodes[(size_t)tok * E_DIM + j];
    if (t == 0) dH[idx] = (_Float16)0.0f;
    else {
        int tok2 = trajs[bt - 1];
        dH[idx] = (_Float16)e_nodes[(size_t)tok2 * E_DIM + j];
    }
}

// ---------------- GRU scan: one block per batch row, 256 threads -----------
// WhhT is the transposed recurrent weight [H][3H]: at step k every lane reads
// WhhT[k*768 + tid] (+256, +512) -> fully coalesced streaming from L2.
__global__ void gru_scan(const float* __restrict__ xp, const float* __restrict__ WhhT,
                         const float* __restrict__ bih, const float* __restrict__ bhh,
                         const float* __restrict__ h0, const int* __restrict__ lengths,
                         float* __restrict__ outs)
{
    __shared__ float hs[H_DIM];
    int b = blockIdx.x, tid = threadIdx.x;
    hs[tid] = h0 ? h0[b * H_DIM + tid] : 0.0f;
    __syncthreads();
    int len = lengths[b];
    float bR = bhh[tid], bZ = bhh[H_DIM + tid], bN = bhh[2 * H_DIM + tid];
    float iR = bih[tid], iZ = bih[H_DIM + tid], iN = bih[2 * H_DIM + tid];
    for (int t = 0; t < T_DIM; ++t) {
        const float* xpt = xp + ((size_t)(b * T_DIM + t)) * H3_DIM;
        float aR = bR, aZ = bZ, aN = bN;
#pragma unroll 4
        for (int k = 0; k < H_DIM; ++k) {
            float hk = hs[k];
            const float* wp = WhhT + (size_t)k * H3_DIM;
            aR += hk * wp[tid];
            aZ += hk * wp[H_DIM + tid];
            aN += hk * wp[2 * H_DIM + tid];
        }
        float xr = xpt[tid] + iR;
        float xz = xpt[H_DIM + tid] + iZ;
        float xn = xpt[2 * H_DIM + tid] + iN;
        float r = 1.0f / (1.0f + __expf(-(xr + aR)));
        float z = 1.0f / (1.0f + __expf(-(xz + aZ)));
        float n = tanhf(xn + r * aN);
        float hnew = (1.0f - z) * n + z * hs[tid];
        __syncthreads();
        hs[tid] = hnew;
        __syncthreads();
        outs[((size_t)(b * T_DIM + t)) * H_DIM + tid] = (t < len) ? hnew : 0.0f;
    }
}

// ---------------- attention score per (b,t) --------------------------------
__global__ void attn_score(const float* __restrict__ outs, const float* __restrict__ w_omega,
                           const float* __restrict__ u_omega, const int* __restrict__ lengths,
                           float* __restrict__ att)
{
    __shared__ float row[H_DIM];
    __shared__ float red[H_DIM];
    int bt = blockIdx.x, tid = threadIdx.x;
    int b = bt / T_DIM, t = bt % T_DIM;
    row[tid] = outs[(size_t)bt * H_DIM + tid];
    __syncthreads();
    float acc = 0.0f;
    for (int h = 0; h < H_DIM; ++h) acc += row[h] * w_omega[(size_t)h * H_DIM + tid];
    red[tid] = tanhf(acc) * u_omega[tid];
    __syncthreads();
    for (int s = 128; s > 0; s >>= 1) { if (tid < s) red[tid] += red[tid + s]; __syncthreads(); }
    if (tid == 0) att[bt] = (t < lengths[b]) ? red[0] : -1e10f;
}

// ---------------- masked softmax over T + weighted pooling -----------------
__global__ void softmax_pool(const float* __restrict__ att, const float* __restrict__ outs,
                             float* __restrict__ encFinal)
{
    __shared__ float sc[T_DIM];
    __shared__ float red[T_DIM];
    int b = blockIdx.x, tid = threadIdx.x;
    float a = att[b * T_DIM + tid];
    red[tid] = a; __syncthreads();
    for (int s = 128; s > 0; s >>= 1) { if (tid < s) red[tid] = fmaxf(red[tid], red[tid + s]); __syncthreads(); }
    float mx = red[0];
    __syncthreads();
    float e = __expf(a - mx);
    red[tid] = e; __syncthreads();
    for (int s = 128; s > 0; s >>= 1) { if (tid < s) red[tid] += red[tid + s]; __syncthreads(); }
    float sum = red[0];
    sc[tid] = e / sum;
    __syncthreads();
    float acc = 0.0f;
    for (int t = 0; t < T_DIM; ++t)
        acc += outs[((size_t)(b * T_DIM + t)) * H_DIM + tid] * sc[t];
    encFinal[b * H_DIM + tid] = acc;
}

// ---------------- mu / logvar / z ------------------------------------------
__global__ void latent(const float* __restrict__ encFinal,
                       const float* __restrict__ muW, const float* __restrict__ mub,
                       const float* __restrict__ lvW, const float* __restrict__ lvb,
                       const float* __restrict__ eps,
                       float* __restrict__ muOut, float* __restrict__ lvOut,
                       float* __restrict__ zOut, float* __restrict__ zWs)
{
    __shared__ float ef[H_DIM];
    int b = blockIdx.x, tid = threadIdx.x;
    ef[tid] = encFinal[b * H_DIM + tid];
    __syncthreads();
    float m = mub[tid], l = lvb[tid];
    const float* mr = muW + (size_t)tid * H_DIM;
    const float* lr = lvW + (size_t)tid * H_DIM;
    for (int k = 0; k < H_DIM; ++k) { float v = ef[k]; m += v * mr[k]; l += v * lr[k]; }
    float zv = m + eps[b * H_DIM + tid] * __expf(0.5f * l);
    muOut[b * H_DIM + tid] = m;
    lvOut[b * H_DIM + tid] = l;
    zOut [b * H_DIM + tid] = zv;
    zWs  [b * H_DIM + tid] = zv;
}

// ---------------- layernorm -> f16 -----------------------------------------
__global__ void layernorm_f16(const float* __restrict__ dec, const float* __restrict__ g,
                              const float* __restrict__ bb, _Float16* __restrict__ lnH)
{
    __shared__ float red[H_DIM];
    int bt = blockIdx.x, tid = threadIdx.x;
    float x = dec[(size_t)bt * H_DIM + tid];
    red[tid] = x; __syncthreads();
    for (int s = 128; s > 0; s >>= 1) { if (tid < s) red[tid] += red[tid + s]; __syncthreads(); }
    float mean = red[0] * (1.0f / H_DIM);
    __syncthreads();
    float d = x - mean;
    red[tid] = d * d; __syncthreads();
    for (int s = 128; s > 0; s >>= 1) { if (tid < s) red[tid] += red[tid + s]; __syncthreads(); }
    float var = red[0] * (1.0f / H_DIM);
    float y = d * rsqrtf(var + 1e-5f) * g[tid] + bb[tid];
    lnH[(size_t)bt * H_DIM + tid] = (_Float16)y;
}

// ---------------------------------------------------------------------------
static inline int gemmBlocks(int mTiles, int tilesNJob, int NB) {
    int jobs = mTiles * (tilesNJob / NB);
    return (jobs * 32 + 255) / 256;
}

extern "C" void kernel_launch(void* const* d_in, const int* in_sizes, int n_in,
                              void* d_out, int out_size, void* d_ws, size_t ws_size,
                              hipStream_t stream)
{
    (void)in_sizes; (void)n_in; (void)out_size; (void)ws_size;
    const int*   trajs    = (const int*)d_in[0];
    const int*   lengths  = (const int*)d_in[1];
    const int*   erow     = (const int*)d_in[2];
    const int*   ecol     = (const int*)d_in[3];
    const float* eval     = (const float*)d_in[4];
    const float* d_diag   = (const float*)d_in[5];
    const float* emb      = (const float*)d_in[6];
    const float* Vp       = (const float*)d_in[7];
    const float* enc_Wih  = (const float*)d_in[8];
    const float* enc_Whh  = (const float*)d_in[9];
    const float* enc_bih  = (const float*)d_in[10];
    const float* enc_bhh  = (const float*)d_in[11];
    const float* w_omega  = (const float*)d_in[12];
    const float* u_omega  = (const float*)d_in[13];
    const float* fcmu_W   = (const float*)d_in[14];
    const float* fcmu_b   = (const float*)d_in[15];
    const float* fclv_W   = (const float*)d_in[16];
    const float* fclv_b   = (const float*)d_in[17];
    const float* dec_Wih  = (const float*)d_in[18];
    const float* dec_Whh  = (const float*)d_in[19];
    const float* dec_bih  = (const float*)d_in[20];
    const float* dec_bhh  = (const float*)d_in[21];
    const float* ln_g     = (const float*)d_in[22];
    const float* ln_b     = (const float*)d_in[23];
    const float* out_W    = (const float*)d_in[24];
    const float* out_b    = (const float*)d_in[25];
    const float* eps      = (const float*)d_in[26];

    float* out       = (float*)d_out;                               // [B,T,V]
    float* muOut     = out + (size_t)B_DIM * T_DIM * V_TOK;         // [1,B,H]
    float* lvOut     = muOut + (size_t)B_DIM * H_DIM;
    float* zOut      = lvOut + (size_t)B_DIM * H_DIM;

    // ---- workspace carve-out (4KB aligned chunks) ----
    char* ws = (char*)d_ws;
    size_t off = 0;
    auto carve = [&](size_t bytes) -> char* {
        char* p = ws + off;
        off += (bytes + 4095) & ~(size_t)4095;
        return p;
    };
    float*    agg     = (float*)   carve((size_t)V_TOK * E_DIM * 4);
    float*    e_nodes = (float*)   carve((size_t)V_TOK * E_DIM * 4);
    _Float16* aggH    = (_Float16*)carve((size_t)V_TOK * E_DIM * 2);
    _Float16* outWH   = (_Float16*)carve((size_t)V_TOK * H_DIM * 2);
    float*    xpEnc   = (float*)   carve((size_t)B_DIM * T_DIM * H3_DIM * 4);
    float*    xpDec   = (float*)   carve((size_t)B_DIM * T_DIM * H3_DIM * 4);
    float*    outsEnc = (float*)   carve((size_t)B_DIM * T_DIM * H_DIM * 4);
    float*    outsDec = (float*)   carve((size_t)B_DIM * T_DIM * H_DIM * 4);
    _Float16* lnH     = (_Float16*)carve((size_t)B_DIM * T_DIM * H_DIM * 2);
    _Float16* eH      = (_Float16*)carve((size_t)B_DIM * T_DIM * E_DIM * 2);
    _Float16* dH      = (_Float16*)carve((size_t)B_DIM * T_DIM * E_DIM * 2);
    _Float16* VpTH    = (_Float16*)carve((size_t)E_DIM * E_DIM * 2);
    _Float16* WihEncH = (_Float16*)carve((size_t)H3_DIM * E_DIM * 2);
    _Float16* WihDecH = (_Float16*)carve((size_t)H3_DIM * E_DIM * 2);
    float*    WhhEncT = (float*)   carve((size_t)H3_DIM * H_DIM * 4);
    float*    WhhDecT = (float*)   carve((size_t)H3_DIM * H_DIM * 4);
    float*    att     = (float*)   carve((size_t)B_DIM * T_DIM * 4);
    float*    encF    = (float*)   carve((size_t)B_DIM * H_DIM * 4);
    float*    zWs     = (float*)   carve((size_t)B_DIM * H_DIM * 4);

    // 1) agg = 0 ; scatter-add SpMM
    fill_zero_f32<<<4096, 256, 0, stream>>>(agg, (long)V_TOK * E_DIM);
    spmm_scatter<<<(NE_DIM * (long)E_DIM) / 256, 256, 0, stream>>>(erow, ecol, eval, d_diag, emb, agg);

    // 2) operand conversions / transposes
    cvt_f32_to_f16<<<((long)V_TOK * E_DIM + 255) / 256, 256, 0, stream>>>(agg, aggH, (long)V_TOK * E_DIM);
    transpose_vp_f16<<<(E_DIM * E_DIM) / 256, 256, 0, stream>>>(Vp, VpTH);
    cvt_f32_to_f16<<<((long)H3_DIM * E_DIM + 255) / 256, 256, 0, stream>>>(enc_Wih, WihEncH, (long)H3_DIM * E_DIM);
    cvt_f32_to_f16<<<((long)H3_DIM * E_DIM + 255) / 256, 256, 0, stream>>>(dec_Wih, WihDecH, (long)H3_DIM * E_DIM);
    cvt_f32_to_f16<<<((long)V_TOK * H_DIM + 255) / 256, 256, 0, stream>>>(out_W, outWH, (long)V_TOK * H_DIM);
    transpose_f32<<<((long)H3_DIM * H_DIM + 255) / 256, 256, 0, stream>>>(enc_Whh, WhhEncT, H3_DIM, H_DIM);
    transpose_f32<<<((long)H3_DIM * H_DIM + 255) / 256, 256, 0, stream>>>(dec_Whh, WhhDecT, H3_DIM, H_DIM);

    // 3) e_nodes = agg @ Vp   (WMMA: 20000 x 128 x 128, 8 N-tiles -> NB=4)
    wmma_gemm_ab_t<4><<<gemmBlocks(V_TOK / 16, E_DIM / 16, 4), 256, 0, stream>>>(
        aggH, VpTH, nullptr, e_nodes, V_TOK, E_DIM, E_DIM, 0, E_DIM / 16, 0);

    // 4) gather encoder input + shifted decoder input (f16)
    gather_inputs<<<((long)B_DIM * T_DIM * E_DIM) / 256, 256, 0, stream>>>(e_nodes, trajs, eH, dH);

    // 5) encoder xp = e_input @ Wih^T  (WMMA: 4096 x 768 x 128, 48 N-tiles)
    wmma_gemm_ab_t<4><<<gemmBlocks(B_DIM * T_DIM / 16, H3_DIM / 16, 4), 256, 0, stream>>>(
        eH, WihEncH, nullptr, xpEnc, B_DIM * T_DIM, H3_DIM, E_DIM, 0, H3_DIM / 16, 0);

    // 6) encoder GRU scan (one WG per batch row, coalesced WhhT stream)
    gru_scan<<<B_DIM, H_DIM, 0, stream>>>(xpEnc, WhhEncT, enc_bih, enc_bhh, nullptr, lengths, outsEnc);

    // 7) attention + masked softmax pooling
    attn_score<<<B_DIM * T_DIM, H_DIM, 0, stream>>>(outsEnc, w_omega, u_omega, lengths, att);
    softmax_pool<<<B_DIM, T_DIM, 0, stream>>>(att, outsEnc, encF);

    // 8) mu / logvar / z
    latent<<<B_DIM, H_DIM, 0, stream>>>(encF, fcmu_W, fcmu_b, fclv_W, fclv_b, eps,
                                        muOut, lvOut, zOut, zWs);

    // 9) decoder xp = d_input @ Wih^T
    wmma_gemm_ab_t<4><<<gemmBlocks(B_DIM * T_DIM / 16, H3_DIM / 16, 4), 256, 0, stream>>>(
        dH, WihDecH, nullptr, xpDec, B_DIM * T_DIM, H3_DIM, E_DIM, 0, H3_DIM / 16, 0);

    // 10) decoder GRU scan, h0 = z
    gru_scan<<<B_DIM, H_DIM, 0, stream>>>(xpDec, WhhDecT, dec_bih, dec_bhh, zWs, lengths, outsDec);

    // 11) layernorm -> f16
    layernorm_f16<<<B_DIM * T_DIM, H_DIM, 0, stream>>>(outsDec, ln_g, ln_b, lnH);

    // 12) output = ln @ out_W^T + out_b  (WMMA: 4096 x 20000 x 256)
    //     1250 N-tiles = 312 jobs of 4 tiles + one tail pass of 2 tiles.
    {
        const int tilesN = V_TOK / 16;            // 1250
        const int mainTiles = (tilesN / 4) * 4;   // 1248
        const int tailTiles = tilesN - mainTiles; // 2
        wmma_gemm_ab_t<4><<<gemmBlocks(B_DIM * T_DIM / 16, mainTiles, 4), 256, 0, stream>>>(
            lnH, outWH, out_b, out, B_DIM * T_DIM, V_TOK, H_DIM, 0, mainTiles, 1);
        wmma_gemm_ab_t<2><<<gemmBlocks(B_DIM * T_DIM / 16, tailTiles, 2), 256, 0, stream>>>(
            lnH, outWH, out_b, out, B_DIM * T_DIM, V_TOK, H_DIM, mainTiles * 16, tailTiles, 1);
    }
}
